// MultiHeadAttentionPacked_39668317946507
// MI455X (gfx1250) — compile-verified, matrix operand order checked by reading, not run on hardware
//
#include <hip/hip_runtime.h>
#include <hip/hip_bf16.h>

#define SEQ    2048
#define BATCH  4
#define DMODEL 1024
#define NHEAD  16
#define DK     64

typedef __bf16 v16bf __attribute__((ext_vector_type(16)));
typedef float  v8f   __attribute__((ext_vector_type(8)));

union Frag {
  uint4          u[2];
  unsigned short s[16];
  v16bf          v;
};
static_assert(sizeof(Frag) == 32, "frag size");

__device__ __forceinline__ unsigned short f2bf(float x) {
  unsigned u = __float_as_uint(x);
  u += 0x7FFFu + ((u >> 16) & 1u);      // round-to-nearest-even
  return (unsigned short)(u >> 16);
}

__device__ __forceinline__ v8f vzero8() {
  v8f z = {0.f, 0.f, 0.f, 0.f, 0.f, 0.f, 0.f, 0.f};
  return z;
}

// 16x32 bf16 WMMA fragment loader.  For A: lane = row; for B: lane = column.
// base points at (tile_row0/col0, k0); ld = element stride between rows/cols.
// Lane<16: K chunks {0..7},{16..23}; lane>=16: {8..15},{24..31}.
__device__ __forceinline__ void load_frag_bf16(Frag& f, const unsigned short* base,
                                               int ld, int lane) {
  const int r    = lane & 15;
  const int half = lane >> 4;
  const unsigned short* p = base + r * ld + half * 8;
  f.u[0] = *reinterpret_cast<const uint4*>(p);
  f.u[1] = *reinterpret_cast<const uint4*>(p + 16);
}

__device__ __forceinline__ v8f wmma_bf16(const Frag& a, const Frag& b, v8f c) {
  return __builtin_amdgcn_wmma_f32_16x16x32_bf16(false, a.v, false, b.v,
                                                 (short)0, c, false, false);
}

// ---------------------------------------------------------------------------
// Kernel 1: bulk fp32 -> bf16 conversion.  Straight copies for Wqk, Wv and
// the three activation tensors (vectorized 4-wide); transposed gather for
// Wo2 (wo2t[e,d] = Wvot[1024+d, e]) so its B-fragments are contiguous.
// One thread = 4 elements.  Unit ranges:
//   [0,        524288)  wqk_b   (2M elems)
//   [524288,   786432)  wv_b    (1M)
//   [786432,  1048576)  wo2t    (1M, transposed)
//   [1048576, 3145728)  xq_b    (8M)
//   [3145728, 5242880)  xk_b    (8M)
//   [5242880, 7340032)  xv_b    (8M)
// ---------------------------------------------------------------------------
__device__ __forceinline__ void cvt4(const float* __restrict__ s,
                                     unsigned short* __restrict__ d) {
  float4 v = *reinterpret_cast<const float4*>(s);
  unsigned lo = (unsigned)f2bf(v.x) | ((unsigned)f2bf(v.y) << 16);
  unsigned hi = (unsigned)f2bf(v.z) | ((unsigned)f2bf(v.w) << 16);
  uint2 o; o.x = lo; o.y = hi;
  *reinterpret_cast<uint2*>(d) = o;
}

__global__ void cvt_all(const float* __restrict__ wqk, const float* __restrict__ wvot,
                        const float* __restrict__ xq, const float* __restrict__ xk,
                        const float* __restrict__ xv,
                        unsigned short* __restrict__ wqk_b,
                        unsigned short* __restrict__ wv_b,
                        unsigned short* __restrict__ wo2t,
                        unsigned short* __restrict__ xqb,
                        unsigned short* __restrict__ xkb,
                        unsigned short* __restrict__ xvb) {
  size_t t = (size_t)blockIdx.x * 256 + threadIdx.x;   // unit index (4 elems)
  if (t < 524288) {
    size_t j = t * 4;
    cvt4(wqk + j, wqk_b + j);
  } else if (t < 786432) {
    size_t j = (t - 524288) * 4;
    cvt4(wvot + j, wv_b + j);
  } else if (t < 1048576) {
    size_t j = (t - 786432) * 4;                       // index into wo2t [e,d]
    int e = (int)(j >> 10), d = (int)(j & 1023);
#pragma unroll
    for (int m = 0; m < 4; ++m)
      wo2t[(size_t)e * DMODEL + d + m] =
          f2bf(wvot[(size_t)(DMODEL + d + m) * DMODEL + e]);
  } else if (t < 3145728) {
    size_t j = (t - 1048576) * 4;
    cvt4(xq + j, xqb + j);
  } else if (t < 5242880) {
    size_t j = (t - 3145728) * 4;
    cvt4(xk + j, xkb + j);
  } else {
    size_t j = (t - 5242880) * 4;
    cvt4(xv + j, xvb + j);
  }
}

// ---------------------------------------------------------------------------
// Kernel 2: fused Q/K/V projection GEMM (all-bf16 operands, fp32 accum).
// z=0:Q, z=1:K, z=2:V.  M = SEQ*BATCH (m = s*BATCH+b), N = K = DMODEL.
// Q,K written bf16 [b,h,s,dk]; V written transposed [b,h,dk,s].
// Block = 256 thr (8 waves); block tile 32(M) x 256(N); wave tile 16 x 64.
// Double-buffered K-loop.
// ---------------------------------------------------------------------------
__global__ void proj_qkv(const unsigned short* __restrict__ xqb,
                         const unsigned short* __restrict__ xkb,
                         const unsigned short* __restrict__ xvb,
                         const unsigned short* __restrict__ wqk_b,
                         const unsigned short* __restrict__ wv_b,
                         unsigned short* __restrict__ qb,
                         unsigned short* __restrict__ kb,
                         unsigned short* __restrict__ vt) {
  const int lane = threadIdx.x & 31;
  const int w    = threadIdx.x >> 5;           // 0..7
  const int z    = blockIdx.z;
  const unsigned short* xsrc = (z == 0) ? xqb : (z == 1) ? xkb : xvb;
  const unsigned short* wsrc =
      (z == 0) ? wqk_b : (z == 1) ? (wqk_b + (size_t)DMODEL * DMODEL) : wv_b;

  const int m0 = blockIdx.x * 32 + (w & 1) * 16;
  const int n0 = blockIdx.y * 256 + (w >> 1) * 64;

  v8f acc[4] = {vzero8(), vzero8(), vzero8(), vzero8()};

  Frag a_cur, b_cur[4];
  load_frag_bf16(a_cur, xsrc + (size_t)m0 * DMODEL, DMODEL, lane);
#pragma unroll
  for (int t = 0; t < 4; ++t)
    load_frag_bf16(b_cur[t], wsrc + (size_t)(n0 + t * 16) * DMODEL, DMODEL, lane);

#pragma unroll 2
  for (int kk = 0; kk < DMODEL; kk += 32) {
    const int kn = (kk + 32 < DMODEL) ? kk + 32 : 0;   // wrap: harmless load
    Frag a_nxt, b_nxt[4];
    load_frag_bf16(a_nxt, xsrc + (size_t)m0 * DMODEL + kn, DMODEL, lane);
#pragma unroll
    for (int t = 0; t < 4; ++t)
      load_frag_bf16(b_nxt[t], wsrc + (size_t)(n0 + t * 16) * DMODEL + kn,
                     DMODEL, lane);
#pragma unroll
    for (int t = 0; t < 4; ++t)
      acc[t] = wmma_bf16(a_cur, b_cur[t], acc[t]);
    a_cur = a_nxt;
#pragma unroll
    for (int t = 0; t < 4; ++t) b_cur[t] = b_nxt[t];
  }

  const int r0 = (lane >> 4) * 8;
  const int c  = lane & 15;
#pragma unroll
  for (int t = 0; t < 4; ++t) {
#pragma unroll
    for (int i = 0; i < 8; ++i) {
      int m = m0 + r0 + i;
      int n = n0 + t * 16 + c;
      int s = m >> 2, b = m & 3;                 // m = s*BATCH + b
      int h = n >> 6, dk = n & 63;
      unsigned short val = f2bf(acc[t][i]);
      if (z == 0)
        qb[(((size_t)(b * NHEAD + h) * SEQ + s) * DK) + dk] = val;
      else if (z == 1)
        kb[(((size_t)(b * NHEAD + h) * SEQ + s) * DK) + dk] = val;
      else
        vt[(((size_t)(b * NHEAD + h) * DK + dk) * SEQ) + s] = val;
    }
  }
}

// ---------------------------------------------------------------------------
// Kernel 3: attention.  One wave owns (b, h, 16 q-rows).
// Pass 1: stream K in steps of 32: scores via WMMA -> exp -> row-sum accum +
//         un-normalized context via WMMA (probabilities round-trip through a
//         per-wave LDS tile to convert C-layout -> A-layout).  K fragments
//         for the next tile are issued before the exp/LDS/context section so
//         global latency hides behind transcendental + DS + WMMA work.
// Pass 2: recompute exp(scores), emit normalized p/NHEAD into attn_avg with
//         fp32 global atomics (64 MB target is L2-resident).
// ---------------------------------------------------------------------------
__global__ void attn_fwd(const unsigned short* __restrict__ qb,
                         const unsigned short* __restrict__ kb,
                         const unsigned short* __restrict__ vt,
                         unsigned short* __restrict__ ctx,
                         float* __restrict__ attn_avg) {
  __shared__ __align__(16) unsigned short esh[4][16 * 32];   // 4 KB

  const int lane = threadIdx.x & 31;
  const int wl   = threadIdx.x >> 5;
  const int wg   = blockIdx.x * 4 + wl;        // 0 .. 8191
  const int qt = wg & 127;                     // SEQ/16 = 128
  const int h  = (wg >> 7) & 15;
  const int b  = wg >> 11;
  const int q0 = qt * 16;

  const unsigned short* qbase = qb + ((size_t)(b * NHEAD + h) * SEQ + q0) * DK;
  const unsigned short* kbase = kb + (size_t)(b * NHEAD + h) * SEQ * DK;
  const unsigned short* vbase = vt + (size_t)(b * NHEAD + h) * DK * SEQ;
  unsigned short* my = esh[wl];

  Frag aq0, aq1;
  load_frag_bf16(aq0, qbase + 0,  DK, lane);   // d = 0..31
  load_frag_bf16(aq1, qbase + 32, DK, lane);   // d = 32..63

  v8f ctxacc[4] = {vzero8(), vzero8(), vzero8(), vzero8()};
  v8f rowsum = vzero8();

  const int r0 = (lane >> 4) * 8;
  const int c  = lane & 15;
  const float scale = 0.125f;                  // 1/sqrt(DK)

  Frag k00, k01, k10, k11;                     // current K-tile fragments
  load_frag_bf16(k00, kbase + 0,       DK, lane);
  load_frag_bf16(k01, kbase + 32,      DK, lane);
  load_frag_bf16(k10, kbase + 16 * DK, DK, lane);
  load_frag_bf16(k11, kbase + 16 * DK + 32, DK, lane);

#pragma unroll 2
  for (int kk = 0; kk < SEQ; kk += 32) {
    v8f s0 = vzero8(), s1 = vzero8();
    s0 = wmma_bf16(aq0, k00, s0);
    s0 = wmma_bf16(aq1, k01, s0);
    s1 = wmma_bf16(aq0, k10, s1);
    s1 = wmma_bf16(aq1, k11, s1);

    // Prefetch next K tile while exp/LDS/context work proceeds.
    const int kn = (kk + 32 < SEQ) ? kk + 32 : 0;
    Frag n00, n01, n10, n11;
    load_frag_bf16(n00, kbase + (size_t)kn * DK,             DK, lane);
    load_frag_bf16(n01, kbase + (size_t)kn * DK + 32,        DK, lane);
    load_frag_bf16(n10, kbase + (size_t)(kn + 16) * DK,      DK, lane);
    load_frag_bf16(n11, kbase + (size_t)(kn + 16) * DK + 32, DK, lane);

#pragma unroll
    for (int i = 0; i < 8; ++i) {
      float e0 = __expf(s0[i] * scale);
      float e1 = __expf(s1[i] * scale);
      rowsum[i] += e0 + e1;
      my[(r0 + i) * 32 + c]      = f2bf(e0);
      my[(r0 + i) * 32 + 16 + c] = f2bf(e1);
    }
    asm volatile("s_wait_dscnt 0" ::: "memory");

    Frag ep;
    load_frag_bf16(ep, my, 32, lane);          // 16x32 probability tile
#pragma unroll
    for (int t = 0; t < 4; ++t) {
      Frag bv;
      load_frag_bf16(bv, vbase + (size_t)(t * 16) * SEQ + kk, SEQ, lane);
      ctxacc[t] = wmma_bf16(ep, bv, ctxacc[t]);
    }
    k00 = n00; k01 = n01; k10 = n10; k11 = n11;
  }

  // Row-sum reduction across the 16-lane halves (rows 0-7 live in lanes 0-15,
  // rows 8-15 in lanes 16-31; xor offsets < 16 keep halves separate).
  v8f inv;
#pragma unroll
  for (int i = 0; i < 8; ++i) {
    float r = rowsum[i];
    r += __shfl_xor(r, 1, 32);
    r += __shfl_xor(r, 2, 32);
    r += __shfl_xor(r, 4, 32);
    r += __shfl_xor(r, 8, 32);
    inv[i] = 1.0f / r;
  }

  // Normalize context and store bf16 at [m = s*BATCH+b, h*64+dk].
#pragma unroll
  for (int t = 0; t < 4; ++t) {
#pragma unroll
    for (int i = 0; i < 8; ++i) {
      int s = q0 + r0 + i;
      int col = h * DK + t * 16 + c;
      ctx[(size_t)(s * BATCH + b) * DMODEL + col] = f2bf(ctxacc[t][i] * inv[i]);
    }
  }

  // Pass 2: normalized probabilities into attn_avg (mean over heads).
  float* abase = attn_avg + (size_t)b * SEQ * SEQ + (size_t)q0 * SEQ;
  const float hs = 1.0f / (float)NHEAD;
  Frag p0, p1;
  load_frag_bf16(p0, kbase + 0,  DK, lane);
  load_frag_bf16(p1, kbase + 32, DK, lane);
#pragma unroll 2
  for (int kk = 0; kk < SEQ; kk += 16) {
    v8f s0 = vzero8();
    s0 = wmma_bf16(aq0, p0, s0);
    s0 = wmma_bf16(aq1, p1, s0);
    const int kn = (kk + 16 < SEQ) ? kk + 16 : 0;
    Frag q0f, q1f;
    load_frag_bf16(q0f, kbase + (size_t)kn * DK,      DK, lane);
    load_frag_bf16(q1f, kbase + (size_t)kn * DK + 32, DK, lane);
#pragma unroll
    for (int i = 0; i < 8; ++i) {
      float p = __expf(s0[i] * scale) * inv[i] * hs;
      atomicAdd(abase + (size_t)(r0 + i) * SEQ + kk + c, p);
    }
    p0 = q0f; p1 = q1f;
  }
}

// ---------------------------------------------------------------------------
// Kernel 4: output projection.  out[m, e] = sum_d ctx[m, d] * wo2t[e, d].
// Double-buffered like proj_qkv; fp32 result straight to d_out.
// ---------------------------------------------------------------------------
__global__ void out_proj(const unsigned short* __restrict__ ctx,
                         const unsigned short* __restrict__ wo2t,
                         float* __restrict__ out) {
  const int lane = threadIdx.x & 31;
  const int w    = threadIdx.x >> 5;
  const int m0 = blockIdx.x * 32 + (w & 1) * 16;
  const int n0 = blockIdx.y * 256 + (w >> 1) * 64;

  v8f acc[4] = {vzero8(), vzero8(), vzero8(), vzero8()};

  Frag a_cur, b_cur[4];
  load_frag_bf16(a_cur, ctx + (size_t)m0 * DMODEL, DMODEL, lane);
#pragma unroll
  for (int t = 0; t < 4; ++t)
    load_frag_bf16(b_cur[t], wo2t + (size_t)(n0 + t * 16) * DMODEL, DMODEL, lane);

#pragma unroll 2
  for (int kk = 0; kk < DMODEL; kk += 32) {
    const int kn = (kk + 32 < DMODEL) ? kk + 32 : 0;
    Frag a_nxt, b_nxt[4];
    load_frag_bf16(a_nxt, ctx + (size_t)m0 * DMODEL + kn, DMODEL, lane);
#pragma unroll
    for (int t = 0; t < 4; ++t)
      load_frag_bf16(b_nxt[t], wo2t + (size_t)(n0 + t * 16) * DMODEL + kn,
                     DMODEL, lane);
#pragma unroll
    for (int t = 0; t < 4; ++t)
      acc[t] = wmma_bf16(a_cur, b_cur[t], acc[t]);
    a_cur = a_nxt;
#pragma unroll
    for (int t = 0; t < 4; ++t) b_cur[t] = b_nxt[t];
  }

  const int r0 = (lane >> 4) * 8;
  const int c  = lane & 15;
#pragma unroll
  for (int t = 0; t < 4; ++t) {
#pragma unroll
    for (int i = 0; i < 8; ++i) {
      out[(size_t)(m0 + r0 + i) * DMODEL + n0 + t * 16 + c] = acc[t][i];
    }
  }
}

// ---------------------------------------------------------------------------
extern "C" void kernel_launch(void* const* d_in, const int* in_sizes, int n_in,
                              void* d_out, int out_size, void* d_ws, size_t ws_size,
                              hipStream_t stream) {
  (void)in_sizes; (void)n_in; (void)out_size; (void)ws_size;
  const float* q    = (const float*)d_in[0];
  const float* k    = (const float*)d_in[1];
  const float* v    = (const float*)d_in[2];
  const float* wqk  = (const float*)d_in[3];
  const float* wvot = (const float*)d_in[4];

  float* out      = (float*)d_out;
  float* attn_avg = out + (size_t)SEQ * BATCH * DMODEL;

  char* w = (char*)d_ws;
  unsigned short* wqk_b = (unsigned short*)w; w += (size_t)2 * DMODEL * DMODEL * 2;
  unsigned short* wv_b  = (unsigned short*)w; w += (size_t)DMODEL * DMODEL * 2;
  unsigned short* wo2t  = (unsigned short*)w; w += (size_t)DMODEL * DMODEL * 2;
  unsigned short* xqb   = (unsigned short*)w; w += (size_t)SEQ * BATCH * DMODEL * 2;
  unsigned short* xkb   = (unsigned short*)w; w += (size_t)SEQ * BATCH * DMODEL * 2;
  unsigned short* xvb   = (unsigned short*)w; w += (size_t)SEQ * BATCH * DMODEL * 2;
  unsigned short* qb    = (unsigned short*)w; w += (size_t)BATCH * NHEAD * SEQ * DK * 2;
  unsigned short* kb    = (unsigned short*)w; w += (size_t)BATCH * NHEAD * SEQ * DK * 2;
  unsigned short* vt    = (unsigned short*)w; w += (size_t)BATCH * NHEAD * SEQ * DK * 2;
  unsigned short* ctx   = (unsigned short*)w; w += (size_t)SEQ * BATCH * DMODEL * 2;

  hipMemsetAsync(attn_avg, 0, (size_t)BATCH * SEQ * SEQ * sizeof(float), stream);

  // 7,340,032 four-element units / 256 threads = 28,672 blocks.
  cvt_all<<<28672, 256, 0, stream>>>(wqk, wvot, q, k, v,
                                     wqk_b, wv_b, wo2t, xqb, xkb, xvb);
  proj_qkv<<<dim3(256, 4, 3), 256, 0, stream>>>(xqb, xkb, xvb, wqk_b, wv_b,
                                                qb, kb, vt);
  attn_fwd<<<2048, 128, 0, stream>>>(qb, kb, vt, ctx, attn_avg);
  out_proj<<<dim3(256, 4), 256, 0, stream>>>(ctx, wo2t, out);
}
